// RNN_1726576853837
// MI455X (gfx1250) — matrix operand order repaired and checked by reading, not compiled, not run on hardware
//
#include <hip/hip_runtime.h>
#include <hip/hip_bf16.h>
#include <math.h>

#define T_STEPS 512
#define BATCH   64
#define DIN     512
#define HID     512
#define G4      2048   // 4*HID
#define OUTW    1024
#define NSLICE  32     // blocks per direction (H/16)

typedef __attribute__((ext_vector_type(16))) __bf16       v16bf;
typedef __attribute__((ext_vector_type(8)))  float        v8f;
typedef __attribute__((ext_vector_type(4)))  unsigned int u32x4;

union Frag { v16bf v; u32x4 q[2]; };

// A-operand (16-bit, 16x32): lane<16 holds row, K {kb..kb+7, kb+16..kb+23};
// lane>=16 same rows, K {kb+8..kb+15, kb+24..kb+31}. Two b128 loads.
__device__ __forceinline__ v16bf load_a_frag(const __bf16* __restrict__ base, int ld,
                                             int row, int kb, int lane) {
    const __bf16* p = base + (size_t)row * ld + kb + ((lane >> 4) << 3);
    Frag f;
    f.q[0] = *(const u32x4*)(p);
    f.q[1] = *(const u32x4*)(p + 16);
    return f.v;
}

// B-operand (16-bit, 32x16) from LDS weight slab row (col-contiguous W row):
// lane<16 -> K {kb..kb+7, kb+16..kb+23}; lane>=16 -> K {kb+8..kb+15, kb+24..kb+31}. Two ds_load_b128.
__device__ __forceinline__ v16bf load_b_frag_lds(const __bf16* row, int kb, int lane) {
    const __bf16* p = row + kb + ((lane >> 4) << 3);
    Frag f;
    f.q[0] = *(const u32x4*)(p);
    f.q[1] = *(const u32x4*)(p + 16);
    return f.v;
}

__device__ __forceinline__ float sigf(float x) { return 1.0f / (1.0f + __expf(-x)); }

__global__ void __launch_bounds__(256) f32_to_bf16_kernel(const float* __restrict__ src,
                                                          __bf16* __restrict__ dst, int n) {
    int i = blockIdx.x * 256 + threadIdx.x;
    if (i < n) dst[i] = (__bf16)src[i];
}

__global__ void __launch_bounds__(256) bias_combine_kernel(const float* __restrict__ bihF,
                                                           const float* __restrict__ bhhF,
                                                           const float* __restrict__ bihB,
                                                           const float* __restrict__ bhhB,
                                                           float* __restrict__ bias) {
    int i = blockIdx.x * 256 + threadIdx.x;
    if (i < G4) {
        bias[i]      = bihF[i] + bhhF[i];
        bias[G4 + i] = bihB[i] + bhhB[i];
    }
}

__global__ void __launch_bounds__(256) zero_state_kernel(__bf16* __restrict__ h,
                                                         unsigned* __restrict__ bar) {
    int i = blockIdx.x * 256 + threadIdx.x;
    if (i < 2 * 2 * BATCH * HID) h[i] = (__bf16)0.0f;
    if (i < 2)                   bar[i] = 0u;
}

// Persistent bidirectional LSTM. grid = 64 blocks (dir[2] x slice[32]) x 512 threads
// (16 waves: 4 batch-row tiles x 4 gates). Weights stationary in LDS, cell state and
// biases register-resident; per-direction grid barrier between timesteps.
__global__ void __launch_bounds__(512, 1) lstm_persistent_kernel(
        const __bf16* __restrict__ xb,    // [T][B][D] bf16
        const __bf16* __restrict__ WihF, const __bf16* __restrict__ WhhF,
        const __bf16* __restrict__ WihB, const __bf16* __restrict__ WhhB,
        const float*  __restrict__ bias,  // [2][4H]
        __bf16*       __restrict__ hbuf,  // [2 dirs][2 pp][B][H] bf16
        float*        __restrict__ out,   // [T][B][OUTW] f32
        unsigned*     __restrict__ bar) { // [2] per-direction barrier counters
    __shared__ __bf16 wlds[64][DIN + HID];   // 128 KB stationary weight slab
    __shared__ float  gbuf[4][BATCH][17];    // padded to avoid bank conflicts

    const int dir   = blockIdx.x >> 5;
    const int slice = blockIdx.x & 31;
    const int j0    = slice * 16;
    const int tid   = threadIdx.x;
    const int lane  = tid & 31;
    const int wave  = tid >> 5;
    const int mt    = wave & 3;
    const int gate  = wave >> 2;

    const __bf16* Wih = dir ? WihB : WihF;
    const __bf16* Whh = dir ? WhhB : WhhF;

    // ---- Stage this block's 64 weight columns (gate-major) into LDS, b128 chunks ----
    for (int idx = tid; idx < 64 * ((DIN + HID) / 8); idx += 512) {
        const int cl = idx >> 7;            // local column 0..63
        const int k  = (idx & 127) << 3;    // element offset 0..1016
        const int n  = (cl >> 4) * HID + j0 + (cl & 15);
        const __bf16* src = (k < DIN) ? (Wih + (size_t)n * DIN + k)
                                      : (Whh + (size_t)n * HID + (k - DIN));
        *(u32x4*)&wlds[cl][k] = *(const u32x4*)src;
    }
    __syncthreads();

    const int arow = mt * 16 + (lane & 15);
    const __bf16* wrow = &wlds[gate * 16 + (lane & 15)][0];
    unsigned* mybar = bar + dir;

    // Fixed element ownership for the cell phase: column j for rows bb0 and bb0+32.
    const int bb0 = tid >> 4;          // 0..31
    const int jl  = tid & 15;
    const int j   = j0 + jl;
    const float* bs = bias + dir * G4;
    const float bi = bs[0 * HID + j];  // loop-invariant gate biases -> registers
    const float bf = bs[1 * HID + j];
    const float bg = bs[2 * HID + j];
    const float bo = bs[3 * HID + j];
    float c0 = 0.0f, c1 = 0.0f;        // register-resident cell state

    for (int t = 0; t < T_STEPS; ++t) {
        const int t_in = dir ? (T_STEPS - 1 - t) : t;
        const __bf16* x_t  = xb + (size_t)t_in * BATCH * DIN;
        const __bf16* h_in = hbuf + (size_t)(dir * 2 + (t & 1)) * BATCH * HID;
        __bf16* h_out      = hbuf + (size_t)(dir * 2 + ((t + 1) & 1)) * BATCH * HID;

        // Prefetch next step's x slice into caches (hide HBM latency off the chain).
        if (slice == 0 && t + 1 < T_STEPS) {
            const int tn = dir ? (T_STEPS - 2 - t) : (t + 1);
            const char* pf = (const char*)(xb + (size_t)tn * BATCH * DIN) + (size_t)tid * 128;
            __builtin_prefetch(pf, 0, 1);
        }

        v8f acc = {0.f, 0.f, 0.f, 0.f, 0.f, 0.f, 0.f, 0.f};
        #pragma unroll 4
        for (int kb = 0; kb < DIN; kb += 32) {        // x_t @ Wih^T
            v16bf a = load_a_frag(x_t, DIN, arow, kb, lane);
            v16bf b = load_b_frag_lds(wrow, kb, lane);
            acc = __builtin_amdgcn_wmma_f32_16x16x32_bf16(false, a, false, b,
                                                          (short)0, acc, false, false);
        }
        #pragma unroll 4
        for (int kb = 0; kb < HID; kb += 32) {        // h @ Whh^T
            v16bf a = load_a_frag(h_in, HID, arow, kb, lane);
            v16bf b = load_b_frag_lds(wrow, DIN + kb, lane);
            acc = __builtin_amdgcn_wmma_f32_16x16x32_bf16(false, a, false, b,
                                                          (short)0, acc, false, false);
        }

        // C/D layout: VGPR r -> (M = r + (lane<16?0:8), N = lane&15)
        {
            const int nl    = lane & 15;
            const int mbase = mt * 16 + ((lane >> 4) << 3);
            #pragma unroll
            for (int r = 0; r < 8; ++r)
                gbuf[gate][mbase + r][nl] = acc[r];
        }
        __syncthreads();

        // Fused LSTM cell: each thread owns (bb0, j) and (bb0+32, j); c in registers.
        {
            const float ig0 = gbuf[0][bb0][jl] + bi;
            const float fg0 = gbuf[1][bb0][jl] + bf;
            const float gg0 = gbuf[2][bb0][jl] + bg;
            const float og0 = gbuf[3][bb0][jl] + bo;
            c0 = sigf(fg0) * c0 + sigf(ig0) * tanhf(gg0);
            const float hn0 = sigf(og0) * tanhf(c0);

            const int bb1 = bb0 + 32;
            const float ig1 = gbuf[0][bb1][jl] + bi;
            const float fg1 = gbuf[1][bb1][jl] + bf;
            const float gg1 = gbuf[2][bb1][jl] + bg;
            const float og1 = gbuf[3][bb1][jl] + bo;
            c1 = sigf(fg1) * c1 + sigf(ig1) * tanhf(gg1);
            const float hn1 = sigf(og1) * tanhf(c1);

            h_out[(size_t)bb0 * HID + j] = (__bf16)hn0;
            h_out[(size_t)bb1 * HID + j] = (__bf16)hn1;
            float* orow = out + (size_t)t_in * BATCH * OUTW + (size_t)dir * HID + j;
            orow[(size_t)bb0 * OUTW] = hn0;
            orow[(size_t)bb1 * OUTW] = hn1;
        }

        // ---- Per-direction grid barrier (release h_out, acquire peers' h_out) ----
        __threadfence();          // release: make this thread's stores device-visible
        __syncthreads();
        if (tid == 0) {
            __hip_atomic_fetch_add(mybar, 1u, __ATOMIC_RELEASE, __HIP_MEMORY_SCOPE_AGENT);
            while (__hip_atomic_load(mybar, __ATOMIC_ACQUIRE, __HIP_MEMORY_SCOPE_AGENT)
                   < (unsigned)NSLICE * (unsigned)(t + 1)) {
                __builtin_amdgcn_s_sleep(1);
            }
        }
        __syncthreads();
        __threadfence();          // acquire: invalidate L0 on every CU of this block
    }
}

extern "C" void kernel_launch(void* const* d_in, const int* in_sizes, int n_in,
                              void* d_out, int out_size, void* d_ws, size_t ws_size,
                              hipStream_t stream) {
    const float* x    = (const float*)d_in[0];
    const float* WihF = (const float*)d_in[1];
    const float* WhhF = (const float*)d_in[2];
    const float* bihF = (const float*)d_in[3];
    const float* bhhF = (const float*)d_in[4];
    const float* WihB = (const float*)d_in[5];
    const float* WhhB = (const float*)d_in[6];
    const float* bihB = (const float*)d_in[7];
    const float* bhhB = (const float*)d_in[8];
    float* out = (float*)d_out;

    char* ws = (char*)d_ws;
    size_t off = 0;
    auto carve = [&](size_t bytes) -> void* {
        void* p = ws + off;
        off = (off + bytes + 255) & ~(size_t)255;
        return p;
    };
    __bf16*   xb    = (__bf16*)carve((size_t)T_STEPS * BATCH * DIN * 2);  // 32 MB
    __bf16*   WihFb = (__bf16*)carve((size_t)G4 * DIN * 2);
    __bf16*   WhhFb = (__bf16*)carve((size_t)G4 * HID * 2);
    __bf16*   WihBb = (__bf16*)carve((size_t)G4 * DIN * 2);
    __bf16*   WhhBb = (__bf16*)carve((size_t)G4 * HID * 2);
    float*    bias  = (float*)carve((size_t)2 * G4 * 4);
    __bf16*   hbuf  = (__bf16*)carve((size_t)2 * 2 * BATCH * HID * 2);
    unsigned* bar   = (unsigned*)carve(256);

    const int nx = T_STEPS * BATCH * DIN;
    const int nw = G4 * DIN;
    f32_to_bf16_kernel<<<(nx + 255) / 256, 256, 0, stream>>>(x, xb, nx);
    f32_to_bf16_kernel<<<(nw + 255) / 256, 256, 0, stream>>>(WihF, WihFb, nw);
    f32_to_bf16_kernel<<<(nw + 255) / 256, 256, 0, stream>>>(WhhF, WhhFb, nw);
    f32_to_bf16_kernel<<<(nw + 255) / 256, 256, 0, stream>>>(WihB, WihBb, nw);
    f32_to_bf16_kernel<<<(nw + 255) / 256, 256, 0, stream>>>(WhhB, WhhBb, nw);
    bias_combine_kernel<<<(G4 + 255) / 256, 256, 0, stream>>>(bihF, bhhF, bihB, bhhB, bias);
    zero_state_kernel<<<(2 * 2 * BATCH * HID + 255) / 256, 256, 0, stream>>>(hbuf, bar);

    lstm_persistent_kernel<<<2 * NSLICE, 512, 0, stream>>>(xb, WihFb, WhhFb, WihBb, WhhBb,
                                                           bias, hbuf, out, bar);
}
